// HQQLinearLoRA_39530878992971
// MI455X (gfx1250) — matrix operand order repaired and checked by reading, compiled, not run on hardware
//
#include <hip/hip_runtime.h>

// ---------------------------------------------------------------------------
// HQQ 4-bit (int32-held) dequant linear + folded LoRA + bias, f16 WMMA GEMM.
// M = B*S = 8192, N = O = 4096, K = I = 4096, group=64, R=16, scaling=2.0
// GEMM uses double-buffered LDS staged via GLOBAL_LOAD_ASYNC_TO_LDS_B128.
// ---------------------------------------------------------------------------

typedef __attribute__((ext_vector_type(16))) _Float16   v16h;
typedef __attribute__((ext_vector_type(8)))  float      v8f;
typedef __attribute__((ext_vector_type(4)))  unsigned   v4u;
typedef __attribute__((ext_vector_type(4)))  int        v4i;

union Frag16 { v16h h; v4u u[2]; };
union H8     { _Float16 h[8]; v4u u; };

#define TM   128      // block tile M
#define TN   128      // block tile N
#define TK   64       // block tile K (halfs)
#define LDSS 72       // padded LDS row stride in halfs (144B -> bank spread)

#if defined(__HIP_DEVICE_COMPILE__) && __has_builtin(__builtin_amdgcn_global_load_async_to_lds_b128)
#define ASYNC_LDS_CP 1
#endif

// One 16B lane-chunk: global -> LDS.
__device__ __forceinline__ void async_copy16(const _Float16* g, _Float16* l)
{
#ifdef ASYNC_LDS_CP
  // Builtin signature (per compiler diagnostic): param0 = v4i* in AS1
  // (printed as '__device__ *'), non-const; param1 = LDS-side pointer;
  // then imm offset, imm cpol.
  __builtin_amdgcn_global_load_async_to_lds_b128(
      (__attribute__((address_space(1))) v4i*)g,
      (__attribute__((address_space(3))) v4i*)l,
      0, 0);
#else
  *(v4u*)l = *(const v4u*)g;
#endif
}

__device__ __forceinline__ void wait_async_zero()
{
#ifdef ASYNC_LDS_CP
#if __has_builtin(__builtin_amdgcn_s_wait_asynccnt)
  __builtin_amdgcn_s_wait_asynccnt(0);
#else
  asm volatile("s_wait_asynccnt 0x0" ::: "memory");
#endif
#endif
}

// ---------------------------------------------------------------------------
// Kernel 1: W_eff[o,i] = (W_q - zero[o,g]) * scale[o,g] + 2 * sum_r A[i,r]B[r,o]
// ---------------------------------------------------------------------------
__global__ __launch_bounds__(256) void hqq_dequant_fold_kernel(
    const int*   __restrict__ Wq,
    const float* __restrict__ scale,
    const float* __restrict__ zero,
    const float* __restrict__ lA,     // [I,16]
    const float* __restrict__ lB,     // [16,O]
    _Float16*    __restrict__ Wh,     // [O,I] f16 out
    int O, int I, int G)
{
  const int o = blockIdx.x;
  const int t = threadIdx.x;

  float lb[16];
#pragma unroll
  for (int r = 0; r < 16; ++r) lb[r] = lB[r * O + o] * 2.0f;   // fold SCALING

  const int base = t * 16;
  const int g    = base >> 6;                                  // /GROUP_SIZE
  const float sc = scale[o * G + g];
  const float zp = zero [o * G + g];

#pragma unroll 4
  for (int u = 0; u < 16; ++u) {
    const int i = base + u;
    float w = ((float)Wq[(size_t)o * I + i] - zp) * sc;
    float lv = 0.0f;
#pragma unroll
    for (int r = 0; r < 16; ++r) lv = fmaf(lA[i * 16 + r], lb[r], lv);
    Wh[(size_t)o * I + i] = (_Float16)(w + lv);
  }
}

// ---------------------------------------------------------------------------
// Kernel 2: x (f32) -> xh (f16), 8 elements per thread (16B stores)
// ---------------------------------------------------------------------------
__global__ __launch_bounds__(256) void x_to_f16_kernel(
    const float* __restrict__ x, _Float16* __restrict__ xh, size_t n)
{
  size_t idx = ((size_t)blockIdx.x * 256 + threadIdx.x) * 8;
  if (idx + 8 > n) return;
  float4 a = *(const float4*)(x + idx);
  float4 b = *(const float4*)(x + idx + 4);
  H8 o;
  o.h[0] = (_Float16)a.x; o.h[1] = (_Float16)a.y;
  o.h[2] = (_Float16)a.z; o.h[3] = (_Float16)a.w;
  o.h[4] = (_Float16)b.x; o.h[5] = (_Float16)b.y;
  o.h[6] = (_Float16)b.z; o.h[7] = (_Float16)b.w;
  *(v4u*)(xh + idx) = o.u;
}

// ---------------------------------------------------------------------------
// Kernel 3: C[m,n] = sum_k xh[m,k] * Wh[n,k] + bias[n]
// 256 threads = 8 waves; waves in 4(M) x 2(N); each wave: 32x64 via 2x4 WMMAs.
// Double-buffered LDS; async global->LDS staging overlapped with WMMA.
// ---------------------------------------------------------------------------
__global__ __launch_bounds__(256) void wmma_gemm_bias_kernel(
    const _Float16* __restrict__ xh,   // [M,K]
    const _Float16* __restrict__ wh,   // [N,K]
    const float*    __restrict__ bias, // [N]
    float*          __restrict__ out,  // [M,N]
    int M, int N, int K)
{
  __shared__ _Float16 sA[2][TM * LDSS];
  __shared__ _Float16 sB[2][TN * LDSS];

  const int bn   = blockIdx.x * TN;
  const int bm   = blockIdx.y * TM;
  const int tid  = threadIdx.x;
  const int wave = tid >> 5;
  const int lane = tid & 31;
  const int hf   = lane >> 4;        // lane-half (0/1)
  const int l16  = lane & 15;
  const int wm   = (wave & 3) * 32;  // wave M offset in tile
  const int wn   = (wave >> 2) * 64; // wave N offset in tile

  v8f acc[2][4];
#pragma unroll
  for (int i = 0; i < 2; ++i)
#pragma unroll
    for (int j = 0; j < 4; ++j) acc[i][j] = (v8f)(0.0f);

  const _Float16* gA = xh + (size_t)bm * K;   // A tile rows base
  const _Float16* gB = wh + (size_t)bn * K;   // B tile rows base

  // Stage both 128x64 tiles for K-offset kt into LDS buffer `buf`.
  // 1024 16B chunks per matrix, 256 threads -> 4 chunks each.
  auto stage = [&](int kt, int buf) {
#pragma unroll
    for (int it = 0; it < 4; ++it) {
      const int c   = tid + it * 256;
      const int row = c >> 3;
      const int col = (c & 7) * 8;
      async_copy16(gA + (size_t)row * K + kt + col, &sA[buf][row * LDSS + col]);
    }
#pragma unroll
    for (int it = 0; it < 4; ++it) {
      const int c   = tid + it * 256;
      const int row = c >> 3;
      const int col = (c & 7) * 8;
      async_copy16(gB + (size_t)row * K + kt + col, &sB[buf][row * LDSS + col]);
    }
  };

  const int nk = K / TK;

  // Prologue: stage tile 0.
  stage(0, 0);
  wait_async_zero();
  __syncthreads();

  for (int t = 0; t < nk; ++t) {
    const int cur = t & 1;

    // Issue async staging of the NEXT tile into the other buffer; it lands
    // while this iteration's WMMAs run. Safe: the buffer being filled was
    // last read in iteration t-1, whose reads drained at the last barrier
    // (s_wait_dscnt 0 precedes barrier-signal).
    if (t + 1 < nk) stage((t + 1) * TK, cur ^ 1);

    const _Float16* lA_ = sA[cur];
    const _Float16* lB_ = sB[cur];

#pragma unroll
    for (int kk = 0; kk < TK; kk += 32) {
      Frag16 a[2], b[4];
      // A fragment (16x32 f16): row m = l16; elements 0..7 -> K=kk+hf*8+e,
      // elements 8..15 -> K=kk+16+hf*8+(e-8).
#pragma unroll
      for (int i = 0; i < 2; ++i) {
        const int row = wm + i * 16 + l16;
        a[i].u[0] = *(const v4u*)(lA_ + row * LDSS + kk + hf * 8);
        a[i].u[1] = *(const v4u*)(lA_ + row * LDSS + kk + 16 + hf * 8);
      }
      // B fragment (32x16 f16): lane column n = l16; elements 0..7 -> K=kk+hf*8+e,
      // elements 8..15 -> K=kk+16+hf*8+(e-8).
#pragma unroll
      for (int j = 0; j < 4; ++j) {
        const int col = wn + j * 16 + l16;
        b[j].u[0] = *(const v4u*)(lB_ + col * LDSS + kk + hf * 8);
        b[j].u[1] = *(const v4u*)(lB_ + col * LDSS + kk + 16 + hf * 8);
      }
#pragma unroll
      for (int i = 0; i < 2; ++i)
#pragma unroll
        for (int j = 0; j < 4; ++j)
          acc[i][j] = __builtin_amdgcn_wmma_f32_16x16x32_f16(
              false, a[i].h, false, b[j].h,
              (short)0, acc[i][j], false, false);
    }

    // Next tile's async writes must have landed before anyone reads them;
    // this wave's LDS reads must drain before the buffer is refilled (the
    // compiler's s_wait_dscnt-before-barrier handles that half).
    wait_async_zero();
    __syncthreads();
  }

  // ---- epilogue: C/D layout: VGPR r -> M = r + 8*hf, N = l16 -------------
#pragma unroll
  for (int j = 0; j < 4; ++j) {
    const int n = bn + wn + j * 16 + l16;
    const float bv = bias[n];
#pragma unroll
    for (int i = 0; i < 2; ++i) {
      const int mbase = bm + wm + i * 16 + hf * 8;
#pragma unroll
      for (int r = 0; r < 8; ++r)
        out[(size_t)(mbase + r) * N + n] = acc[i][j][r] + bv;
    }
  }
}

// ---------------------------------------------------------------------------
extern "C" void kernel_launch(void* const* d_in, const int* in_sizes, int n_in,
                              void* d_out, int out_size, void* d_ws, size_t ws_size,
                              hipStream_t stream)
{
  const float* x     = (const float*)d_in[0];
  const int*   Wq    = (const int*)  d_in[1];
  const float* scale = (const float*)d_in[2];
  const float* zero  = (const float*)d_in[3];
  const float* lA    = (const float*)d_in[4];
  const float* lB    = (const float*)d_in[5];
  const float* bias  = (const float*)d_in[6];
  float*       out   = (float*)d_out;

  const int I = 4096, O = 4096, G = 64;
  const int M = in_sizes[0] / I;          // B*S = 8192

  _Float16* xh = (_Float16*)d_ws;
  _Float16* Wh = (_Float16*)((char*)d_ws + (size_t)M * I * sizeof(_Float16));

  // 1) dequant + LoRA fold -> Wh [O,I]
  hqq_dequant_fold_kernel<<<O, 256, 0, stream>>>(Wq, scale, zero, lA, lB, Wh, O, I, G);

  // 2) x -> f16
  const size_t nx = (size_t)M * I;
  x_to_f16_kernel<<<(unsigned)(nx / 8 / 256), 256, 0, stream>>>(x, xh, nx);

  // 3) WMMA GEMM + bias
  dim3 grid(O / TN, M / TM);
  wmma_gemm_bias_kernel<<<grid, 256, 0, stream>>>(xh, Wh, bias, out, M, O, I);
}
